// SmallPredictor_25520695673247
// MI455X (gfx1250) — matrix operand order repaired and checked by reading, not compiled, not run on hardware
//
#include <hip/hip_runtime.h>

typedef __attribute__((ext_vector_type(16))) _Float16 v16h;
typedef __attribute__((ext_vector_type(8)))  _Float16 v8h;
typedef __attribute__((ext_vector_type(8)))  float    v8f;

#define H_    128
#define W_    128
#define HW    16384     // 128*128
#define BATCH 64
#define CST   24        // channel slots per pixel in global buffers (channel-minor)
#define PST   40        // channel slots per pixel in LDS (bank-conflict-free stride)
#define WPL   34        // LDS tile width (32 + 2 halo cols)

// ---------------------------------------------------------------------------
// Build concat buffer C once per call: slots 0..11 = 0 (rewritten by layers
// 2/5/8), slots 12..21 = x (f32->f16), slots 22..23 = 0.  Channel-minor.
// ---------------------------------------------------------------------------
__global__ void __launch_bounds__(256) init_concat(const float* __restrict__ x,
                                                   _Float16* __restrict__ C) {
    size_t p = (size_t)blockIdx.x * 256 + threadIdx.x;   // 0 .. BATCH*HW-1
    int s   = (int)(p >> 14);
    int pix = (int)(p & 16383);
    const float* xp = x + (size_t)s * 10 * HW + pix;
    _Float16 h[10];
#pragma unroll
    for (int c = 0; c < 10; c++) h[c] = (_Float16)xp[(size_t)c * HW];
    v8h g0 = {};                                        // slots 0..7
    v8h g1 = {};                                        // slots 8..15
    v8h g2 = {};                                        // slots 16..23
    g1[4] = h[0]; g1[5] = h[1]; g1[6] = h[2]; g1[7] = h[3];     // slots 12..15
    g2[0] = h[4]; g2[1] = h[5]; g2[2] = h[6]; g2[3] = h[7];     // slots 16..19
    g2[4] = h[8]; g2[5] = h[9];                                  // slots 20..21
    _Float16* d = C + p * CST;
    *(v8h*)(d)      = g0;
    *(v8h*)(d + 8)  = g1;
    *(v8h*)(d + 16) = g2;
}

// ---------------------------------------------------------------------------
// Per-sample implicit-GEMM conv (KSxKS, SAME) + optional BN + ReLU via
// v_wmma_f32_16x16x32_f16, tap-major / channel-minor K ordering.
//  - block = 16 rows x 32 cols of one sample, one 16-wide out-channel tile
//  - 8 waves = 2 x-groups x 4 y-groups; each wave owns 4 row-tiles (T=4)
//  - K chunk ck = tap ck (dy=ck/3, dx=ck%3), 32 channel slots per chunk
//  - A fragment: one padded-row LDS read; B fragment: ONE contiguous 32B read
//    at pixel_base + tap_offset + kgrp*16  (2 x ds_load_b128, conflict-free)
//  - padded K slots have zero weights, so their B values are don't-care
// ---------------------------------------------------------------------------
template<int KS>
__global__ void __launch_bounds__(256) conv_wmma_kernel(
    const _Float16* __restrict__ in, long in_sstride,
    const float*   __restrict__ wsrc, long w_sstride, int w_off,
    int CINR, int cslot_off, int Cout,
    const float*   __restrict__ bn_gamma, const float* __restrict__ bn_beta,
    const float*   __restrict__ bn_mean,  const float* __restrict__ bn_var, int bn_off,
    _Float16* __restrict__ out16, float* __restrict__ out32, long out_sstride,
    int relu)
{
    constexpr int NCHUNK = (KS == 3) ? 9 : 1;     // one chunk per tap
    constexpr int HALO   = (KS == 3) ? 1 : 0;
    constexpr int ROWS   = 16 + 2 * HALO;         // staged rows
    constexpr int NPIX   = ROWS * WPL;
    constexpr int WROWS  = NCHUNK * 2 * 16;       // A-fragment rows (chunk,kgrp,m)
    constexpr int AP     = 24;                    // padded A row stride (halves)

    __shared__ _Float16 s_in[NPIX * PST];         // activations, channel-minor
    __shared__ _Float16 s_wf[WROWS * AP];         // weights in A-fragment order
    __shared__ float    s_scale[16], s_bias[16];

    const int bx  = blockIdx.x;
    const int x0b = (bx & 3) * 32;
    const int y0b = (bx >> 2) * 16;
    const int mt  = blockIdx.y;
    const int s   = blockIdx.z;
    const int tid = threadIdx.x;

    // ---- phase 1: zero weight fragments, stage activations, BN constants ----
    for (int i = tid; i < WROWS * AP; i += 256) s_wf[i] = (_Float16)0.f;

    if (tid < 16) {
        float sc = 1.f, bi = 0.f;
        int o = mt * 16 + tid;
        if (bn_off >= 0) {
            sc = 0.f; bi = 0.f;
            if (o < Cout) {
                float gm = bn_gamma[bn_off + o];
                float bt = bn_beta [bn_off + o];
                float mn = bn_mean [bn_off + o];
                float vr = bn_var  [bn_off + o];
                sc = gm * rsqrtf(vr + 1e-5f);
                bi = bt - mn * sc;
            }
        }
        s_scale[tid] = sc;
        s_bias [tid] = bi;
    }

    const _Float16* gin = in + (size_t)s * in_sstride;
    for (int i = tid; i < NPIX * 5; i += 256) {   // 5 8-half groups per pixel (2 pad)
        int p  = i / 5, g = i - p * 5;
        int rr = p / WPL, xx = p - rr * WPL;
        int yy = y0b + rr - HALO;
        int xi = x0b + xx - 1;
        v8h v = {};
        if (g < 3 && yy >= 0 && yy < H_ && xi >= 0 && xi < W_)
            v = *(const v8h*)(gin + (size_t)(yy * W_ + xi) * CST + g * 8);
        *(v8h*)&s_in[p * PST + g * 8] = v;
    }
    __syncthreads();

    // ---- phase 2: scatter real weights into fragment slots ----
    // A element j of lane (kgrp): slot kgrp*16+j
    const float* wbase = wsrc + (size_t)s * w_sstride + w_off;   // [Cout][CINR*KS*KS]
    __builtin_prefetch(wbase, 0, 0);
    const int KT = CINR * KS * KS;
    for (int e = tid; e < 16 * KT; e += 256) {
        int m = e / KT, q = e - m * KT;
        int o = mt * 16 + m;
        if (o < Cout) {
            float wv = wbase[o * KT + q];
            int c   = q / (KS * KS);
            int tap = q - c * (KS * KS);
            int r   = c + cslot_off;              // channel slot within chunk
            int kgrp, j;
            kgrp = r >> 4; j = r & 15;
            s_wf[((tap * 2 + kgrp) * 16 + m) * AP + j] = (_Float16)wv;
        }
    }
    __syncthreads();

    // ---- WMMA main loops ----
    const int lane = tid & 31;
    const int wid  = tid >> 5;
    const int xg   = wid & 1;        // x-group (0..1) -> 16 pixels
    const int yg   = wid >> 1;       // y-group (0..3) -> 4 rows
    const int n    = lane & 15;
    const int kgrp = lane >> 4;
    const int kg16 = kgrp * 16;

    int pixbase[4];
#pragma unroll
    for (int t = 0; t < 4; t++)
        pixbase[t] = (((yg * 4 + t) * WPL) + xg * 16 + n) * PST + kg16;

    v8f acc[4] = {};
#pragma unroll
    for (int ck = 0; ck < NCHUNK; ck++) {
        const int dy = (KS == 3) ? (ck / 3) : 0;
        const int dx = (KS == 3) ? (ck - dy * 3) : 0;
        const int tapoff = (KS == 3) ? (dy * WPL + dx) * PST : PST;  // 1x1: col+1

        const int arow = ((ck * 2 + kgrp) * 16 + n) * AP;
        v8h alo = *(const v8h*)&s_wf[arow];
        v8h ahi = *(const v8h*)&s_wf[arow + 8];
        v16h a  = __builtin_shufflevector(alo, ahi,
                    0,1,2,3,4,5,6,7,8,9,10,11,12,13,14,15);
#pragma unroll
        for (int t = 0; t < 4; t++) {
            const _Float16* bp = &s_in[pixbase[t] + tapoff];
            v8h blo = *(const v8h*)bp;
            v8h bhi = *(const v8h*)(bp + 8);
            v16h b  = __builtin_shufflevector(blo, bhi,
                        0,1,2,3,4,5,6,7,8,9,10,11,12,13,14,15);
            acc[t] = __builtin_amdgcn_wmma_f32_16x16x32_f16(
                /*neg_a=*/false, a, /*neg_b=*/false, b,
                /*c_mod=*/(short)0, acc[t], /*reuse_a=*/false, /*reuse_b=*/false);
        }
    }

    // ---- epilogue: BN + ReLU, store (guard o<Cout protects C's x slots) ----
    const int xo = x0b + xg * 16 + n;
#pragma unroll
    for (int t = 0; t < 4; t++) {
        const int yo = y0b + yg * 4 + t;
#pragma unroll
        for (int r = 0; r < 8; r++) {
            int mr = r + kgrp * 8;       // C/D layout: VGPR r -> M = r + 8*(lane>=16)
            int o  = mt * 16 + mr;
            if (o < Cout) {
                float v = acc[t][r] * s_scale[mr] + s_bias[mr];
                if (relu) v = fmaxf(v, 0.f);
                if (out32)
                    out32[(size_t)s * out_sstride + (size_t)o * HW + yo * W_ + xo] = v;
                else
                    out16[(size_t)s * out_sstride + (size_t)(yo * W_ + xo) * CST + o]
                        = (_Float16)v;
            }
        }
    }
}

// ---------------------------------------------------------------------------
// Launch: init concat buffer + 10 conv/BN/ReLU layers.  All intermediates are
// channel-minor f16 in workspace (L2-resident).  Buffer plan:
//   C    : concat buffer (x at slots 12..21; layers 2/5/8 rewrite slots 0..11)
//   bufA : ping   bufB : pong
// ---------------------------------------------------------------------------
extern "C" void kernel_launch(void* const* d_in, const int* in_sizes, int n_in,
                              void* d_out, int out_size, void* d_ws, size_t ws_size,
                              hipStream_t stream) {
    const float* x  = (const float*)d_in[0];
    const float* w1 = (const float*)d_in[1];
    const float* w2 = (const float*)d_in[2];
    const float* w3 = (const float*)d_in[3];
    const float* w4 = (const float*)d_in[4];
    const float* g  = (const float*)d_in[5];
    const float* be = (const float*)d_in[6];
    const float* mu = (const float*)d_in[7];
    const float* va = (const float*)d_in[8];
    float* out = (float*)d_out;

    const size_t BUF = (size_t)BATCH * HW * CST * 2;   // 50.3 MB each
    char* ws = (char*)d_ws;
    _Float16* C    = (_Float16*)ws;
    _Float16* bufA = (_Float16*)(ws + BUF);
    _Float16* bufB = (_Float16*)(ws + 2 * BUF);

    const long SS = (long)HW * CST;    // per-sample stride of channel-minor bufs

    init_concat<<<(BATCH * HW) / 256, 256, 0, stream>>>(x, C);

    dim3 g1(32, 1, BATCH), g2(32, 2, BATCH);

    // ---- stage1 ----
    conv_wmma_kernel<3><<<g1,256,0,stream>>>(C,    SS, w1, 3672,    0, 10, 12, 12,
        g, be, mu, va,   0, bufA, nullptr, SS, 1);
    conv_wmma_kernel<3><<<g1,256,0,stream>>>(bufA, SS, w1, 3672, 1080, 12,  0, 12,
        g, be, mu, va,  12, bufB, nullptr, SS, 1);
    conv_wmma_kernel<3><<<g1,256,0,stream>>>(bufB, SS, w1, 3672, 2376, 12,  0, 12,
        g, be, mu, va,  24, C,    nullptr, SS, 1);
    // ---- stage2 (input = C: cur 0..11, x 12..21) ----
    conv_wmma_kernel<3><<<g2,256,0,stream>>>(C,    SS, w2, 12528,    0, 22, 0, 24,
        g, be, mu, va,  36, bufA, nullptr, SS, 1);
    conv_wmma_kernel<3><<<g2,256,0,stream>>>(bufA, SS, w2, 12528, 4752, 24, 0, 24,
        g, be, mu, va,  60, bufB, nullptr, SS, 1);
    conv_wmma_kernel<3><<<g1,256,0,stream>>>(bufB, SS, w2, 12528, 9936, 24, 0, 12,
        g, be, mu, va,  84, C,    nullptr, SS, 1);
    // ---- stage3 ----
    conv_wmma_kernel<3><<<g2,256,0,stream>>>(C,    SS, w3, 12528,    0, 22, 0, 24,
        g, be, mu, va,  96, bufA, nullptr, SS, 1);
    conv_wmma_kernel<3><<<g2,256,0,stream>>>(bufA, SS, w3, 12528, 4752, 24, 0, 24,
        g, be, mu, va, 120, bufB, nullptr, SS, 1);
    conv_wmma_kernel<3><<<g1,256,0,stream>>>(bufB, SS, w3, 12528, 9936, 24, 0, 12,
        g, be, mu, va, 132, C,    nullptr, SS, 1);
    // ---- stage4: 1x1 conv over C (22 ch), f32 planar out, no BN/ReLU ----
    conv_wmma_kernel<1><<<g1,256,0,stream>>>(C,    SS, w4, 242, 0, 22, 0, 11,
        g, be, mu, va, -1, nullptr, out, 11L * HW, 0);
}
